// SelfAttention_3092376453756
// MI455X (gfx1250) — matrix operand order repaired and checked by reading, not compiled, not run on hardware
//
#include <hip/hip_runtime.h>

// ---------- CDNA5 WMMA types ----------
typedef __attribute__((ext_vector_type(16))) __bf16 bf16x16;
typedef __attribute__((ext_vector_type(8)))  float  v8f;

union Frag {            // 16 bf16 = 32 bytes = two 16B chunks
    bf16x16 v;
    uint4   q[2];
};

// Native conversion: clang lowers fptrunc f32->bf16 to the gfx1250 hw path (RNE).
__device__ __forceinline__ __bf16 f2bf(float f) { return (__bf16)f; }

// Pack two converted bf16 into one dword (enables b32/b64 LDS stores).
__device__ __forceinline__ unsigned pk2bf(float a, float b) {
    unsigned short lo = __builtin_bit_cast(unsigned short, (__bf16)a);
    unsigned short hi = __builtin_bit_cast(unsigned short, (__bf16)b);
    return (unsigned)lo | ((unsigned)hi << 16);
}

#define WMMA_BF16(A, B, C) \
    __builtin_amdgcn_wmma_f32_16x16x32_bf16(false, (A), false, (B), (short)0, (C), false, false)

// =====================================================================
// Kernel 1: fused QKV projection.  C = x @ W + b  -> bf16 out buffers.
// Block tile 128(M) x 64(N), K-step 32. 256 threads = 8 waves (wave32),
// waves arranged 4(M) x 2(N), each wave owns a 32x32 output sub-tile.
// =====================================================================
#define LDSA 40   // A LDS row stride (bf16 elems); 80B = 5*16B
#define LDSB 40   // Bt LDS row stride

__global__ __launch_bounds__(256) void qkv_proj_kernel(
    const float* __restrict__ x,
    const float* __restrict__ Wq, const float* __restrict__ bq,
    const float* __restrict__ Wk, const float* __restrict__ bk,
    const float* __restrict__ Wv, const float* __restrict__ bv,
    __bf16* __restrict__ qo, __bf16* __restrict__ ko, __bf16* __restrict__ vo)
{
    __shared__ __bf16 As[128 * LDSA];      // [m][k]
    __shared__ __bf16 Bt[64 * LDSB];       // transposed: [n][k]

    const float* W; const float* bias; __bf16* out;
    if (blockIdx.z == 0)      { W = Wq; bias = bq; out = qo; }
    else if (blockIdx.z == 1) { W = Wk; bias = bk; out = ko; }
    else                      { W = Wv; bias = bv; out = vo; }

    const int tid  = threadIdx.x;
    const int lane = tid & 31;
    const int wave = tid >> 5;
    const int wm   = (wave & 3) * 32;      // wave M offset within tile
    const int wn   = (wave >> 2) * 32;     // wave N offset within tile
    const int m16  = lane & 15;
    const int hi   = lane >> 4;            // half-wave select
    const int kgA  = hi * 8;               // A-frag K base (ISA layout)

    const int blockM = blockIdx.y * 128;
    const int blockN = blockIdx.x * 64;

    v8f acc[2][2] = {};

    for (int k0 = 0; k0 < 1024; k0 += 32) {
        __syncthreads();
        // ---- stage A: x[blockM..+128][k0..+32] fp32 -> bf16, b64 stores ----
        {
            int r = tid >> 3;              // 0..31
            const int c = (tid & 7) * 4;   // 0..28 (8B aligned in LDS row)
            #pragma unroll
            for (int it = 0; it < 4; ++it) {
                const float4 f = *(const float4*)&x[(size_t)(blockM + r) * 1024 + k0 + c];
                uint2 p;
                p.x = pk2bf(f.x, f.y);
                p.y = pk2bf(f.z, f.w);
                *(uint2*)&As[r * LDSA + c] = p;
                r += 32;
            }
        }
        // ---- stage B transposed: W[k0..+32][blockN..+64] -> Bt[n][k] ----
        {
            int kk = tid >> 4;             // 0..15
            const int n = (tid & 15) * 4;
            #pragma unroll
            for (int it = 0; it < 2; ++it) {
                const float4 f = *(const float4*)&W[(size_t)(k0 + kk) * 1024 + blockN + n];
                Bt[(n + 0) * LDSB + kk] = f2bf(f.x);
                Bt[(n + 1) * LDSB + kk] = f2bf(f.y);
                Bt[(n + 2) * LDSB + kk] = f2bf(f.z);
                Bt[(n + 3) * LDSB + kk] = f2bf(f.w);
                kk += 16;
            }
        }
        __syncthreads();

        Frag a[2], b[2];
        #pragma unroll
        for (int s = 0; s < 2; ++s) {
            const __bf16* ap = &As[(wm + s * 16 + m16) * LDSA + kgA];
            a[s].q[0] = *(const uint4*)(ap);
            a[s].q[1] = *(const uint4*)(ap + 16);
            const __bf16* bp = &Bt[(wn + s * 16 + m16) * LDSB + hi * 8];
            b[s].q[0] = *(const uint4*)(bp);
            b[s].q[1] = *(const uint4*)(bp + 16);
        }
        #pragma unroll
        for (int i = 0; i < 2; ++i)
            #pragma unroll
            for (int j = 0; j < 2; ++j)
                acc[i][j] = WMMA_BF16(a[i].v, b[j].v, acc[i][j]);
    }

    // ---- epilogue: +bias, bf16 store (C layout: VGPR r -> M=r+8*hi, N=lane&15)
    #pragma unroll
    for (int j = 0; j < 2; ++j) {
        const int col = blockN + wn + j * 16 + m16;
        const float bb = bias[col];
        #pragma unroll
        for (int i = 0; i < 2; ++i) {
            const int rowBase = blockM + wm + i * 16 + hi * 8;
            #pragma unroll
            for (int r = 0; r < 8; ++r)
                out[(size_t)(rowBase + r) * 1024 + col] = f2bf(acc[i][j][r] + bb);
        }
    }
}

// =====================================================================
// Kernel 2: flash attention (online softmax, never materialize scores).
// Grid: (32 query-blocks of 128, 16 heads). 8 waves; wave owns 16 rows.
// Key loop: 64 keys/iter. Per iter: 8 WMMA (QK^T) + 8 WMMA (PV).
// =====================================================================
#define KSTR 72   // LDS pitch (bf16 elems); 144B = 9*16B

__global__ __launch_bounds__(256) void flash_attn_kernel(
    const __bf16* __restrict__ qptr, const __bf16* __restrict__ kptr,
    const __bf16* __restrict__ vptr, __bf16* __restrict__ ctxp)
{
    __shared__ __bf16 Ks[64 * KSTR];           // [key][dim]
    __shared__ __bf16 Vt[64 * KSTR];           // transposed: [dim][key]
    __shared__ __bf16 Ps[8 * 16 * KSTR];       // per-wave P staging [16][keys]

    const int tid  = threadIdx.x;
    const int lane = tid & 31;
    const int wave = tid >> 5;
    const int m16  = lane & 15;
    const int hi   = lane >> 4;
    const int kgA  = hi * 8;

    const int head  = blockIdx.y;
    const int cbase = head * 64;
    const int qBase = blockIdx.x * 128 + wave * 16;

    // ---- load Q A-fragments once (rows qBase+m16, dims 0..63) ----
    Frag aQ[2];
    {
        const __bf16* qp = &qptr[(size_t)(qBase + m16) * 1024 + cbase];
        #pragma unroll
        for (int s = 0; s < 2; ++s) {
            aQ[s].q[0] = *(const uint4*)(qp + s * 32 + kgA);
            aQ[s].q[1] = *(const uint4*)(qp + s * 32 + kgA + 16);
        }
    }

    v8f   o[4] = {};                            // 16x64 output accumulator
    float mrow[8], lrow[8];
    #pragma unroll
    for (int r = 0; r < 8; ++r) { mrow[r] = -3.0e38f; lrow[r] = 0.0f; }

    __bf16* pw = &Ps[wave * 16 * KSTR];
    const float scale = 0.125f;                 // 1/sqrt(64)

    for (int kb0 = 0; kb0 < 4096; kb0 += 64) {
        __syncthreads();
        // ---- stage K (row-major) and V (transposed) tiles, bf16 ----
        {
            const int kr = tid >> 2;            // 0..63
            const int c  = (tid & 3) * 16;      // 0..48
            const uint4* ksrc = (const uint4*)&kptr[(size_t)(kb0 + kr) * 1024 + cbase + c];
            *(uint4*)&Ks[kr * KSTR + c]     = ksrc[0];
            *(uint4*)&Ks[kr * KSTR + c + 8] = ksrc[1];

            union { uint4 u[2]; __bf16 el[16]; } vbuf;
            const uint4* vsrc = (const uint4*)&vptr[(size_t)(kb0 + kr) * 1024 + cbase + c];
            vbuf.u[0] = vsrc[0]; vbuf.u[1] = vsrc[1];
            #pragma unroll
            for (int e = 0; e < 16; ++e)
                Vt[(c + e) * KSTR + kr] = vbuf.el[e];

            if (kb0 + 64 < 4096)                // hint next K tile toward L2
                __builtin_prefetch((const char*)&kptr[(size_t)(kb0 + 64 + kr) * 1024 + cbase + c], 0, 1);
        }
        __syncthreads();

        // ---- scores S = Q K^T : 4 key-subtiles x 2 K-steps ----
        v8f s[4];
        #pragma unroll
        for (int j = 0; j < 4; ++j) {
            v8f sj = {};
            #pragma unroll
            for (int t = 0; t < 2; ++t) {
                Frag bK;
                const __bf16* kp = &Ks[(j * 16 + m16) * KSTR + t * 32 + hi * 8];
                bK.q[0] = *(const uint4*)(kp);
                bK.q[1] = *(const uint4*)(kp + 16);
                sj = WMMA_BF16(aQ[t].v, bK.v, sj);
            }
            s[j] = sj * scale;
        }

        // ---- online softmax (row stats via half-wave shfl reductions) ----
        #pragma unroll
        for (int r = 0; r < 8; ++r) {
            float mx = fmaxf(fmaxf(s[0][r], s[1][r]), fmaxf(s[2][r], s[3][r]));
            mx = fmaxf(mx, __shfl_xor(mx, 1, 32));
            mx = fmaxf(mx, __shfl_xor(mx, 2, 32));
            mx = fmaxf(mx, __shfl_xor(mx, 4, 32));
            mx = fmaxf(mx, __shfl_xor(mx, 8, 32));
            const float mnew  = fmaxf(mrow[r], mx);
            const float alpha = __expf(mrow[r] - mnew);
            mrow[r] = mnew;
            float sum = 0.0f;
            #pragma unroll
            for (int j = 0; j < 4; ++j) {
                const float p = __expf(s[j][r] - mnew);
                s[j][r] = p;
                sum += p;
            }
            sum += __shfl_xor(sum, 1, 32);
            sum += __shfl_xor(sum, 2, 32);
            sum += __shfl_xor(sum, 4, 32);
            sum += __shfl_xor(sum, 8, 32);
            lrow[r] = lrow[r] * alpha + sum;
            #pragma unroll
            for (int j = 0; j < 4; ++j) o[j][r] *= alpha;
        }

        // ---- C-layout -> A-layout via per-wave LDS slab (in-order DS) ----
        #pragma unroll
        for (int j = 0; j < 4; ++j)
            #pragma unroll
            for (int r = 0; r < 8; ++r)
                pw[(r + hi * 8) * KSTR + j * 16 + m16] = f2bf(s[j][r]);

        // ---- O += P V : 2 key K-steps x 4 dim-subtiles ----
        #pragma unroll
        for (int t = 0; t < 2; ++t) {
            Frag aP;
            const __bf16* pp = &pw[m16 * KSTR + t * 32 + kgA];
            aP.q[0] = *(const uint4*)(pp);
            aP.q[1] = *(const uint4*)(pp + 16);
            #pragma unroll
            for (int j = 0; j < 4; ++j) {
                Frag bV;
                const __bf16* vp = &Vt[(j * 16 + m16) * KSTR + t * 32 + hi * 8];
                bV.q[0] = *(const uint4*)(vp);
                bV.q[1] = *(const uint4*)(vp + 16);
                o[j] = WMMA_BF16(aP.v, bV.v, o[j]);
            }
        }
    }

    // ---- normalize and store ctx (bf16) ----
    #pragma unroll
    for (int r = 0; r < 8; ++r) {
        const float inv = 1.0f / lrow[r];
        const size_t row = (size_t)(qBase + hi * 8 + r) * 1024 + cbase;
        #pragma unroll
        for (int j = 0; j < 4; ++j)
            ctxp[row + j * 16 + m16] = f2bf(o[j][r] * inv);
    }
}

// =====================================================================
// Kernel 3: output projection.  out = ctx(bf16) @ Wo + bo  (fp32 out)
// =====================================================================
__global__ __launch_bounds__(256) void out_proj_kernel(
    const __bf16* __restrict__ A, const float* __restrict__ W,
    const float* __restrict__ bias, float* __restrict__ out)
{
    __shared__ __bf16 As[128 * LDSA];
    __shared__ __bf16 Bt[64 * LDSB];

    const int tid  = threadIdx.x;
    const int lane = tid & 31;
    const int wave = tid >> 5;
    const int wm   = (wave & 3) * 32;
    const int wn   = (wave >> 2) * 32;
    const int m16  = lane & 15;
    const int hi   = lane >> 4;
    const int kgA  = hi * 8;

    const int blockM = blockIdx.y * 128;
    const int blockN = blockIdx.x * 64;

    v8f acc[2][2] = {};

    for (int k0 = 0; k0 < 1024; k0 += 32) {
        __syncthreads();
        {   // A already bf16: straight 32B copies
            const int r = tid >> 1;            // 0..127
            const int c = (tid & 1) * 16;
            const uint4* src = (const uint4*)&A[(size_t)(blockM + r) * 1024 + k0 + c];
            *(uint4*)&As[r * LDSA + c]     = src[0];
            *(uint4*)&As[r * LDSA + c + 8] = src[1];
        }
        {   // W fp32 -> bf16, transposed
            int kk = tid >> 4;
            const int n = (tid & 15) * 4;
            #pragma unroll
            for (int it = 0; it < 2; ++it) {
                const float4 f = *(const float4*)&W[(size_t)(k0 + kk) * 1024 + blockN + n];
                Bt[(n + 0) * LDSB + kk] = f2bf(f.x);
                Bt[(n + 1) * LDSB + kk] = f2bf(f.y);
                Bt[(n + 2) * LDSB + kk] = f2bf(f.z);
                Bt[(n + 3) * LDSB + kk] = f2bf(f.w);
                kk += 16;
            }
        }
        __syncthreads();

        Frag a[2], b[2];
        #pragma unroll
        for (int s = 0; s < 2; ++s) {
            const __bf16* ap = &As[(wm + s * 16 + m16) * LDSA + kgA];
            a[s].q[0] = *(const uint4*)(ap);
            a[s].q[1] = *(const uint4*)(ap + 16);
            const __bf16* bp = &Bt[(wn + s * 16 + m16) * LDSB + hi * 8];
            b[s].q[0] = *(const uint4*)(bp);
            b[s].q[1] = *(const uint4*)(bp + 16);
        }
        #pragma unroll
        for (int i = 0; i < 2; ++i)
            #pragma unroll
            for (int j = 0; j < 2; ++j)
                acc[i][j] = WMMA_BF16(a[i].v, b[j].v, acc[i][j]);
    }

    #pragma unroll
    for (int j = 0; j < 2; ++j) {
        const int col = blockN + wn + j * 16 + m16;
        const float bb = bias[col];
        #pragma unroll
        for (int i = 0; i < 2; ++i) {
            const int rowBase = blockM + wm + i * 16 + hi * 8;
            #pragma unroll
            for (int r = 0; r < 8; ++r)
                out[(size_t)(rowBase + r) * 1024 + col] = acc[i][j][r] + bb;
        }
    }
}

// =====================================================================
extern "C" void kernel_launch(void* const* d_in, const int* in_sizes, int n_in,
                              void* d_out, int out_size, void* d_ws, size_t ws_size,
                              hipStream_t stream) {
    const float* x  = (const float*)d_in[0];
    const float* Wq = (const float*)d_in[1];
    const float* bq = (const float*)d_in[2];
    const float* Wk = (const float*)d_in[3];
    const float* bk = (const float*)d_in[4];
    const float* Wv = (const float*)d_in[5];
    const float* bv = (const float*)d_in[6];
    const float* Wo = (const float*)d_in[7];
    const float* bo = (const float*)d_in[8];
    float* out = (float*)d_out;

    const size_t NE = (size_t)4096 * 1024;     // elements per activation
    __bf16* qb  = (__bf16*)d_ws;               // 8 MB each, 32 MB total ws
    __bf16* kbf = qb  + NE;
    __bf16* vbf = kbf + NE;
    __bf16* cbf = vbf + NE;

    dim3 gQKV(16, 32, 3);                      // N tiles, M tiles, {q,k,v}
    qkv_proj_kernel<<<gQKV, 256, 0, stream>>>(x, Wq, bq, Wk, bk, Wv, bv, qb, kbf, vbf);

    dim3 gFA(32, 16, 1);                       // query blocks, heads
    flash_attn_kernel<<<gFA, 256, 0, stream>>>(qb, kbf, vbf, cbf);

    dim3 gO(16, 32, 1);
    out_proj_kernel<<<gO, 256, 0, stream>>>(cbf, Wo, bo, out);
}
